// Block_88605175316669
// MI455X (gfx1250) — matrix operand order repaired and checked by reading, compiled, not run on hardware
//
#include <hip/hip_runtime.h>

// ---------------------------------------------------------------------------
// CDNA5 / gfx1250 transformer block. All heavy GEMMs use
// V_WMMA_F32_16X16X32_BF16 (wave32, fp32 accumulate) with double-buffered
// GLOBAL_LOAD_ASYNC_TO_LDS_B128 staging (ASYNCcnt-tracked). LayerNorm /
// softmax / residuals in fp32. bf16 stored as raw unsigned short (RNE).
// ---------------------------------------------------------------------------

typedef __attribute__((ext_vector_type(16))) __bf16 v16bf;
typedef __attribute__((ext_vector_type(8)))  float  v8f;

union FragAB {
  v16bf v;
  uint4 u[2];
};

__device__ __forceinline__ unsigned short f2bf(float f) {
  union { float f; unsigned u; } cv; cv.f = f;
  unsigned r = cv.u + 0x7FFFu + ((cv.u >> 16) & 1u);  // round-to-nearest-even
  return (unsigned short)(r >> 16);
}

__device__ __forceinline__ v8f wmma_bf16(FragAB a, FragAB b, v8f c) {
  // D = A(16x32) * B(32x16) + C, fp32 accum
  return __builtin_amdgcn_wmma_f32_16x16x32_bf16(
      false, a.v, false, b.v, (short)0, c, false, false);
}

// Async global -> LDS, 16 bytes per lane (512B per wave). Tracked by ASYNCcnt.
// LDS destination byte offset = low 32 bits of the generic shared pointer
__device__ __forceinline__ void async_ld_b128(const unsigned short* gptr,
                                              const unsigned short* lptr) {
  unsigned lds_off = (unsigned)(size_t)lptr;
  asm volatile("global_load_async_to_lds_b128 %0, %1, off"
               :: "v"(lds_off), "v"(gptr)
               : "memory");
}

#define WAIT_ASYNCCNT(n) asm volatile("s_wait_asynccnt %0" :: "n"(n) : "memory")

// ---------------------------------------------------------------------------
// Weight repack: wq/wk/wv (H,C,D) fp32 -> Wqkv_t[1536][512] bf16, row n =
// output unit (section*512 + h*64 + d), col c = input channel.
// ---------------------------------------------------------------------------
__global__ __launch_bounds__(256) void conv_wqkv(
    const float* __restrict__ wq, const float* __restrict__ wk,
    const float* __restrict__ wv, unsigned short* __restrict__ out) {
  int idx = blockIdx.x * 256 + threadIdx.x;        // < 1536*512
  if (idx >= 1536 * 512) return;
  int n = idx >> 9;            // 0..1535
  int c = idx & 511;
  int sec = n >> 9;            // 0=q 1=k 2=v
  int nn = n & 511;
  int h = nn >> 6, d = nn & 63;
  const float* src = (sec == 0) ? wq : ((sec == 1) ? wk : wv);
  out[(size_t)n * 512 + c] = f2bf(src[(size_t)h * 512 * 64 + (size_t)c * 64 + d]);
}

// Generic transpose-convert: in (K x N, fp32, row-major) -> out (N x K, bf16)
__global__ __launch_bounds__(256) void conv_transpose(
    const float* __restrict__ in, unsigned short* __restrict__ out,
    int K, int N) {
  int idx = blockIdx.x * 256 + threadIdx.x;
  if (idx >= K * N) return;
  int n = idx / K, k = idx % K;
  out[(size_t)n * K + k] = f2bf(in[(size_t)k * N + n]);
}

// ---------------------------------------------------------------------------
// LayerNorm over last dim (512), fp32 in -> bf16 out. One wave per row.
// ---------------------------------------------------------------------------
__global__ __launch_bounds__(256) void layernorm_bf16(
    const float* __restrict__ X, const float* __restrict__ g,
    const float* __restrict__ b, unsigned short* __restrict__ out) {
  int lane = threadIdx.x & 31, wid = threadIdx.x >> 5;
  int row = blockIdx.x * 8 + wid;
  const float* xr = X + (size_t)row * 512;
  union { float4 f4[4]; float f[16]; } xv, gv, bv;
  const float4* xp = (const float4*)(xr + lane * 16);
  const float4* gp = (const float4*)(g + lane * 16);
  const float4* bp = (const float4*)(b + lane * 16);
#pragma unroll
  for (int i = 0; i < 4; ++i) { xv.f4[i] = xp[i]; gv.f4[i] = gp[i]; bv.f4[i] = bp[i]; }
  float s = 0.f, ss = 0.f;
#pragma unroll
  for (int e = 0; e < 16; ++e) { s += xv.f[e]; ss += xv.f[e] * xv.f[e]; }
#pragma unroll
  for (int m = 16; m >= 1; m >>= 1) { s += __shfl_xor(s, m); ss += __shfl_xor(ss, m); }
  float mean = s * (1.f / 512.f);
  float var = ss * (1.f / 512.f) - mean * mean;
  float rstd = rsqrtf(var + 1e-5f);
  union { unsigned short s16[16]; uint4 q[2]; } ov;
#pragma unroll
  for (int e = 0; e < 16; ++e)
    ov.s16[e] = f2bf((xv.f[e] - mean) * rstd * gv.f[e] + bv.f[e]);
  uint4* op = (uint4*)(out + (size_t)row * 512 + lane * 16);
  op[0] = ov.q[0]; op[1] = ov.q[1];
}

// ---------------------------------------------------------------------------
// Tiled bf16 WMMA GEMM. C[M,N] = A[M,K] * Bt[N,K]^T (Bt stored N-major).
// Block tile 128x128, 8 waves as 2(M) x 4(N), each wave 64x32 = 4x2 WMMA
// tiles. K-loop double-buffered via async global->LDS (ASYNCcnt).
// EPI: 0 = store bf16; 1 = store f32 (acc + bias + residual);
//      2 = store bf16 relu(acc + bias)
// ---------------------------------------------------------------------------
template <int EPI>
__global__ __launch_bounds__(256) void gemm_bf16(
    const unsigned short* __restrict__ A, const unsigned short* __restrict__ Bt,
    void* __restrict__ outp, const float* __restrict__ bias,
    const float* __restrict__ res, int M, int N, int K) {
  __shared__ alignas(16) unsigned short As[2][128][40];  // 32 + 8 pad
  __shared__ alignas(16) unsigned short Bs[2][128][40];

  int tid = threadIdx.x;
  int lane = tid & 31, wid = tid >> 5;
  int m0 = blockIdx.y * 128, n0 = blockIdx.x * 128;
  int waveM = (wid >> 2) * 64, waveN = (wid & 3) * 32;
  int nL = lane & 15, half = lane >> 4;

  // Per-thread async-load assignment: 2 x 16B chunks per matrix per K-step.
  int row0 = tid >> 2;                 // chunk = tid        -> rows 0..63
  int row1 = (tid + 256) >> 2;         // chunk = tid + 256  -> rows 64..127
  int cc = (tid & 3) * 8;

  v8f zero = {};
  v8f acc[4][2];
#pragma unroll
  for (int i = 0; i < 4; ++i)
#pragma unroll
    for (int j = 0; j < 2; ++j) acc[i][j] = zero;

  const unsigned short* Ab = A + (size_t)m0 * K + cc;
  const unsigned short* Bb = Bt + (size_t)n0 * K + cc;

  // Prologue: stage first K-tile into buffer 0 (4 async ops per thread/wave).
  async_ld_b128(Ab + (size_t)row0 * K, &As[0][row0][cc]);
  async_ld_b128(Bb + (size_t)row0 * K, &Bs[0][row0][cc]);
  async_ld_b128(Ab + (size_t)row1 * K, &As[0][row1][cc]);
  async_ld_b128(Bb + (size_t)row1 * K, &Bs[0][row1][cc]);

  int nsteps = K >> 5;
  for (int st = 0; st < nsteps; ++st) {
    int cur = st & 1;
    int k0n = (st + 1) << 5;
    if (st + 1 < nsteps) {
      // Issue next tile into the other buffer, then wait for current tile
      // (4 newly-issued ops may remain outstanding).
      int nb = cur ^ 1;
      async_ld_b128(Ab + (size_t)row0 * K + k0n, &As[nb][row0][cc]);
      async_ld_b128(Bb + (size_t)row0 * K + k0n, &Bs[nb][row0][cc]);
      async_ld_b128(Ab + (size_t)row1 * K + k0n, &As[nb][row1][cc]);
      async_ld_b128(Bb + (size_t)row1 * K + k0n, &Bs[nb][row1][cc]);
      WAIT_ASYNCCNT(4);
    } else {
      WAIT_ASYNCCNT(0);
    }
    __syncthreads();  // every wave has waited for its own portion of `cur`

    FragAB af[4], bf[2];
    // A frag: lane = row M, K split {half*8 .. +7} and {half*8+16 .. +23}
#pragma unroll
    for (int i = 0; i < 4; ++i) {
      const unsigned short* p = &As[cur][waveM + 16 * i + nL][half * 8];
      af[i].u[0] = *(const uint4*)p;
      af[i].u[1] = *(const uint4*)(p + 16);
    }
    // B frag: lane = col N, K split {half*8 .. +7} and {half*8+16 .. +23}
#pragma unroll
    for (int j = 0; j < 2; ++j) {
      const unsigned short* p = &Bs[cur][waveN + 16 * j + nL][half * 8];
      bf[j].u[0] = *(const uint4*)p;
      bf[j].u[1] = *(const uint4*)(p + 16);
    }
#pragma unroll
    for (int i = 0; i < 4; ++i)
#pragma unroll
      for (int j = 0; j < 2; ++j) acc[i][j] = wmma_bf16(af[i], bf[j], acc[i][j]);
    __syncthreads();  // all waves done reading `cur` before it is overwritten
  }

  // Epilogue. C/D layout: lane holds column nL, VGPR r = row (r + 8*half).
#pragma unroll
  for (int i = 0; i < 4; ++i)
#pragma unroll
    for (int j = 0; j < 2; ++j) {
      float* ap = (float*)&acc[i][j];
      int gn = n0 + waveN + 16 * j + nL;
      float bb = (EPI != 0) ? bias[gn] : 0.f;
#pragma unroll
      for (int rr = 0; rr < 8; ++rr) {
        int gm = m0 + waveM + 16 * i + rr + 8 * half;
        size_t idx = (size_t)gm * N + gn;
        float v = ap[rr];
        if (EPI == 0) {
          ((unsigned short*)outp)[idx] = f2bf(v);
        } else if (EPI == 1) {
          ((float*)outp)[idx] = v + bb + res[idx];
        } else {
          v += bb;
          ((unsigned short*)outp)[idx] = f2bf(v > 0.f ? v : 0.f);
        }
      }
    }
}

// ---------------------------------------------------------------------------
// Attention: one wave per (b,h). T=16, D=64.
// qkv layout: [65536 tokens][1536] bf16, cols: q=h*64+d, k=512+h*64+d,
// v=1024+h*64+d. Output obuf: [65536][512] bf16, col = h*64+d.
// ---------------------------------------------------------------------------
__global__ __launch_bounds__(256) void attention_kernel(
    const unsigned short* __restrict__ qkv, unsigned short* __restrict__ obuf) {
  __shared__ float att_s[8][16][16];                     // 8 KB
  __shared__ alignas(16) unsigned short vT[8][64][24];   // 24 KB (pad 16->24)

  int lane = threadIdx.x & 31, wid = threadIdx.x >> 5;
  int bh = blockIdx.x * 8 + wid;
  int b = bh >> 3, h = bh & 7;
  const unsigned short* base = qkv + (size_t)b * 16 * 1536 + h * 64;
  const unsigned short* qb = base;
  const unsigned short* kp = base + 512;
  const unsigned short* vb = base + 1024;
  int nL = lane & 15, half = lane >> 4;

  // Load V (16x64) and transpose into LDS: vT[d][s]
  {
    int s = lane >> 1, d0 = (lane & 1) * 32;
    const unsigned short* vp = vb + (size_t)s * 1536 + d0;
#pragma unroll
    for (int c = 0; c < 4; ++c) {
      uint4 d4 = *(const uint4*)(vp + c * 8);
      const unsigned short* e = (const unsigned short*)&d4;
#pragma unroll
      for (int t = 0; t < 8; ++t) vT[wid][d0 + c * 8 + t][s] = e[t];
    }
  }

  // Q as A-frags (2 frags cover K=d=0..63), K^T as B-frags
  FragAB qa[2], kf[2];
#pragma unroll
  for (int f = 0; f < 2; ++f) {
    const unsigned short* pq = qb + (size_t)nL * 1536 + f * 32 + half * 8;
    qa[f].u[0] = *(const uint4*)pq;
    qa[f].u[1] = *(const uint4*)(pq + 16);
    const unsigned short* pk = kp + (size_t)nL * 1536 + f * 32 + half * 8;
    kf[f].u[0] = *(const uint4*)pk;
    kf[f].u[1] = *(const uint4*)(pk + 16);
  }

  v8f lg = {};
  lg = wmma_bf16(qa[0], kf[0], lg);
  lg = wmma_bf16(qa[1], kf[1], lg);

  // Causal softmax. Lane holds column s=nL; VGPR rr = row t = rr + 8*half.
  float* lp = (float*)&lg;
#pragma unroll
  for (int rr = 0; rr < 8; ++rr) {
    int t = rr + 8 * half;
    float v = (nL <= t) ? lp[rr] * 0.125f : -1e30f;   // scale = D^-0.5
    float m = v;
    m = fmaxf(m, __shfl_xor(m, 8));
    m = fmaxf(m, __shfl_xor(m, 4));
    m = fmaxf(m, __shfl_xor(m, 2));
    m = fmaxf(m, __shfl_xor(m, 1));
    float p = __expf(v - m);
    float sum = p;
    sum += __shfl_xor(sum, 8);
    sum += __shfl_xor(sum, 4);
    sum += __shfl_xor(sum, 2);
    sum += __shfl_xor(sum, 1);
    att_s[wid][t][nL] = p / sum;
  }
  __syncthreads();

  // att (16x16) as A-frag with K padded to 32 (upper half zero)
  FragAB af;
  {
    union { unsigned short s16[16]; uint4 q[2]; } pk2;
#pragma unroll
    for (int e = 0; e < 16; ++e) pk2.s16[e] = half ? (unsigned short)0 : f2bf(att_s[wid][nL][e]);
    af.u[0] = pk2.q[0];
    af.u[1] = pk2.q[1];
  }

  // o = att @ V : 4 tiles of 16 columns
#pragma unroll
  for (int j = 0; j < 4; ++j) {
    FragAB bfrag;
    if (lane < 16) {  // K=s in 0..15 valid; lanes 16..31 cover padded K=16..31
      const unsigned short* p = &vT[wid][j * 16 + nL][0];
      bfrag.u[0] = *(const uint4*)p;
      bfrag.u[1] = *(const uint4*)(p + 8);
    } else {
      uint4 z = {0, 0, 0, 0};
      bfrag.u[0] = z; bfrag.u[1] = z;
    }
    v8f oc = {};
    oc = wmma_bf16(af, bfrag, oc);
    float* op = (float*)&oc;
    int gn = h * 64 + j * 16 + nL;
#pragma unroll
    for (int rr = 0; rr < 8; ++rr) {
      int gt = b * 16 + rr + 8 * half;
      obuf[(size_t)gt * 512 + gn] = f2bf(op[rr]);
    }
  }
}

// ---------------------------------------------------------------------------
// Host launch sequence.
// Workspace layout (peak ~456 MB):
//   [0, 6MB)      weights bf16 (wqkv_t 1.5M, wproj_t 0.5M, w1_t 2M, w2_t 2M)
//   [8, 72MB)     h / h2 bf16 (64 MB)
//   [72, 264MB)   qkv bf16 (192 MB)     } reused as a1 (256 MB) after attention
//   [264, 328MB)  o bf16 (64 MB)        } and proj complete
//   [328, 456MB)  y fp32 (128 MB)
// ---------------------------------------------------------------------------
extern "C" void kernel_launch(void* const* d_in, const int* in_sizes, int n_in,
                              void* d_out, int out_size, void* d_ws, size_t ws_size,
                              hipStream_t stream) {
  const float* x      = (const float*)d_in[0];
  const float* ln1_g  = (const float*)d_in[1];
  const float* ln1_b  = (const float*)d_in[2];
  const float* wq     = (const float*)d_in[3];
  const float* wk     = (const float*)d_in[4];
  const float* wv     = (const float*)d_in[5];
  const float* w_proj = (const float*)d_in[6];
  const float* b_proj = (const float*)d_in[7];
  const float* ln2_g  = (const float*)d_in[8];
  const float* ln2_b  = (const float*)d_in[9];
  const float* w1     = (const float*)d_in[10];
  const float* b1     = (const float*)d_in[11];
  const float* w2     = (const float*)d_in[12];
  const float* b2     = (const float*)d_in[13];

  const int M = 65536;  // B*T = 4096*16
  char* ws = (char*)d_ws;
  unsigned short* wqkv_t  = (unsigned short*)(ws);
  unsigned short* wproj_t = wqkv_t + (size_t)1536 * 512;
  unsigned short* w1_t    = wproj_t + (size_t)512 * 512;
  unsigned short* w2_t    = w1_t + (size_t)2048 * 512;
  unsigned short* hbuf = (unsigned short*)(ws + (8ull << 20));
  unsigned short* qkv  = (unsigned short*)(ws + (72ull << 20));
  unsigned short* obuf = (unsigned short*)(ws + (264ull << 20));
  unsigned short* a1   = qkv;  // reuses [72MB, 328MB)
  float* ybuf          = (float*)(ws + (328ull << 20));
  float* outF          = (float*)d_out;

  // Weight repack (fp32 -> bf16, N-major)
  conv_wqkv<<<(1536 * 512) / 256, 256, 0, stream>>>(wq, wk, wv, wqkv_t);
  conv_transpose<<<(512 * 512) / 256, 256, 0, stream>>>(w_proj, wproj_t, 512, 512);
  conv_transpose<<<(512 * 2048) / 256, 256, 0, stream>>>(w1, w1_t, 512, 2048);
  conv_transpose<<<(2048 * 512) / 256, 256, 0, stream>>>(w2, w2_t, 2048, 512);

  // LN1
  layernorm_bf16<<<M / 8, 256, 0, stream>>>(x, ln1_g, ln1_b, hbuf);

  // QKV projection: [M,512] x [512,1536] -> bf16
  gemm_bf16<0><<<dim3(1536 / 128, M / 128), 256, 0, stream>>>(
      hbuf, wqkv_t, qkv, nullptr, nullptr, M, 1536, 512);

  // Attention (one wave per (b,h); 4096 blocks x 8 waves)
  attention_kernel<<<4096, 256, 0, stream>>>(qkv, obuf);

  // Output projection + bias + residual(x) -> y (fp32)
  gemm_bf16<1><<<dim3(512 / 128, M / 128), 256, 0, stream>>>(
      obuf, wproj_t, ybuf, b_proj, x, M, 512, 512);

  // LN2 -> h2 (reuse hbuf)
  layernorm_bf16<<<M / 8, 256, 0, stream>>>(ybuf, ln2_g, ln2_b, hbuf);

  // MLP1: relu(h2 @ w1 + b1) -> bf16
  gemm_bf16<2><<<dim3(2048 / 128, M / 128), 256, 0, stream>>>(
      hbuf, w1_t, a1, b1, nullptr, M, 2048, 512);

  // MLP2: a1 @ w2 + b2 + y -> out (fp32)
  gemm_bf16<1><<<dim3(512 / 128, M / 128), 256, 0, stream>>>(
      a1, w2_t, outF, b2, ybuf, M, 512, 2048);
}
